// MultiheadAttention_38972533244165
// MI455X (gfx1250) — matrix operand order repaired and checked by reading, not compiled, not run on hardware
//
#include <hip/hip_runtime.h>
#include <hip/hip_bf16.h>

// ---------------------------------------------------------------------------
// MI455X multi-head attention (fp32 in/out, bf16 WMMA compute)
//   K1: QKV projection  (x@W + b, Q pre-scaled by 1/sqrt(64)) -> bf16 [bh][t][d]
//   K2: flash attention (online softmax, scores never hit HBM) -> bf16 [b,t,h*d]
//   K3: output projection (attn@Wo + bo)                       -> fp32 d_out
// ---------------------------------------------------------------------------

typedef __bf16 bf16;
typedef __attribute__((ext_vector_type(16))) __bf16 v16bf;
typedef __attribute__((ext_vector_type(8)))  __bf16 v8bf;
typedef __attribute__((ext_vector_type(4)))  __bf16 v4bf;
typedef __attribute__((ext_vector_type(8)))  float  v8f;
typedef __attribute__((ext_vector_type(4)))  float  v4f;

#define D_MODEL 1024
#define NHEAD   16
#define HDIM    64
#define SEQ     2048
#define BATCH   2
#define ROWS    (BATCH*SEQ)   // 4096

static __device__ __forceinline__ v16bf make_frag(v8bf lo, v8bf hi) {
    v16bf r;
#pragma unroll
    for (int i = 0; i < 8; ++i) { r[i] = lo[i]; r[i + 8] = hi[i]; }
    return r;
}

// ---------------------------------------------------------------------------
// Kernel 1: fused QKV projection.  blockIdx.z in {0,1,2} selects Q/K/V.
// Tile: 128 rows x 64 cols per 256-thread (8-wave) workgroup.
// Output layout: qkv[z][b*16+h][t][d]  (bf16), Q scaled by 0.125.
// ---------------------------------------------------------------------------
__global__ __launch_bounds__(256)
void qkv_gemm_kernel(const float* __restrict__ x,
                     const float* __restrict__ Wq, const float* __restrict__ bq,
                     const float* __restrict__ Wk, const float* __restrict__ bk,
                     const float* __restrict__ Wv, const float* __restrict__ bv,
                     bf16* __restrict__ qkv_ws)
{
    const int z = blockIdx.z;
    const float* W    = (z == 0) ? Wq : ((z == 1) ? Wk : Wv);
    const float* bias = (z == 0) ? bq : ((z == 1) ? bk : bv);
    bf16* outw = qkv_ws + (size_t)z * ROWS * D_MODEL;

    __shared__ bf16 As[128 * 40];   // x tile   [128 rows][32 k], stride 40 (80B, 16B-aligned)
    __shared__ bf16 Bt[64 * 40];    // W tile transposed [64 n][32 k]

    const int tid    = threadIdx.x;
    const int lane   = tid & 31;
    const int wv_id  = tid >> 5;          // wave 0..7 -> 16 M-rows each
    const int mBlock = blockIdx.y * 128;
    const int nBase  = blockIdx.x * 64;
    const int lh     = lane & 15;
    const int hi16   = lane >> 4;         // 0 or 1 (lane half)
    const int aK0    = hi16 * 8;          // A-frag K chunk base
    const int bK0    = hi16 * 8;          // B-frag K chunk base

    v8f acc[4] = {};

    for (int k0 = 0; k0 < D_MODEL; k0 += 32) {
        // ---- stage A: x[mBlock..+128][k0..+32] fp32 -> bf16 LDS -------------
#pragma unroll
        for (int i = 0; i < 4; ++i) {
            int idx = tid + i * 256;                  // 0..1023
            int r = idx >> 3, cc = idx & 7;
            v4f f = *(const v4f*)(x + (size_t)(mBlock + r) * D_MODEL + k0 + cc * 4);
            v4bf h;
#pragma unroll
            for (int j = 0; j < 4; ++j) h[j] = (bf16)f[j];
            *(v4bf*)(&As[r * 40 + cc * 4]) = h;
        }
        // ---- stage B transposed: W[k0..+32][nBase..+64] -> Bt[n][k] ---------
#pragma unroll
        for (int i = 0; i < 2; ++i) {
            int idx = tid + i * 256;                  // 0..511
            int kr = idx >> 4, nc = idx & 15;
            v4f f = *(const v4f*)(W + (size_t)(k0 + kr) * D_MODEL + nBase + nc * 4);
#pragma unroll
            for (int j = 0; j < 4; ++j) Bt[(nc * 4 + j) * 40 + kr] = (bf16)f[j];
        }
        __syncthreads();

        // ---- WMMA: one A frag, 4 B frags per wave ---------------------------
        const int arow = wv_id * 16 + lh;
        v8bf alo = *(const v8bf*)(&As[arow * 40 + aK0]);
        v8bf ahi = *(const v8bf*)(&As[arow * 40 + aK0 + 16]);
        v16bf afrag = make_frag(alo, ahi);
#pragma unroll
        for (int nt = 0; nt < 4; ++nt) {
            const int brow = nt * 16 + lh;
            v8bf blo = *(const v8bf*)(&Bt[brow * 40 + bK0]);
            v8bf bhi = *(const v8bf*)(&Bt[brow * 40 + bK0 + 16]);
            v16bf bfrag = make_frag(blo, bhi);
            acc[nt] = __builtin_amdgcn_wmma_f32_16x16x32_bf16(
                false, afrag, false, bfrag, (short)0, acc[nt], false, false);
        }
        __syncthreads();
    }

    // ---- epilogue: +bias, scale Q, scatter to [bh][t][d] bf16 ---------------
    const float scale = (z == 0) ? 0.125f : 1.0f;   // 1/sqrt(head_dim) folded into Q
#pragma unroll
    for (int nt = 0; nt < 4; ++nt) {
        const int c = nBase + nt * 16 + lh;
        const float bval = bias[c];
        const int h = c >> 6, d = c & 63;
#pragma unroll
        for (int r = 0; r < 8; ++r) {
            const int m = mBlock + wv_id * 16 + hi16 * 8 + r;
            const int bb = m >> 11;        // batch
            const int t  = m & 2047;       // seq pos
            const float val = (acc[nt][r] + bval) * scale;
            outw[(((size_t)(bb * NHEAD + h) * SEQ + t) * HDIM) + d] = (bf16)val;
        }
    }
}

// ---------------------------------------------------------------------------
// Kernel 2: flash attention.  grid = (SEQ/64 q-tiles, BATCH*NHEAD).
// 128 threads = 4 waves; each wave owns 16 query rows; loops 64-key tiles.
// ---------------------------------------------------------------------------
__global__ __launch_bounds__(128)
void flash_attn_kernel(const bf16* __restrict__ Qw, const bf16* __restrict__ Kw,
                       const bf16* __restrict__ Vw, bf16* __restrict__ attn)
{
    __shared__ bf16 Qs[64 * 72];        // [q][d]   stride 72 (144B, 16B-aligned)
    __shared__ bf16 Ks[64 * 72];        // [key][d]
    __shared__ bf16 Vt[64 * 72];        // [d][key] (transposed for P@V B-frags)
    __shared__ bf16 Ps[4][16 * 72];     // per-wave probs staging [q][key]

    const int tid   = threadIdx.x;
    const int lane  = tid & 31;
    const int wv_id = tid >> 5;         // 0..3
    const int lh    = lane & 15;
    const int hi16  = lane >> 4;
    const int bh    = blockIdx.y;
    const int q0    = blockIdx.x * 64;

    const bf16* Qg = Qw + ((size_t)bh * SEQ + q0) * HDIM;
    const bf16* Kg = Kw + (size_t)bh * SEQ * HDIM;
    const bf16* Vg = Vw + (size_t)bh * SEQ * HDIM;

    // load Q tile 64x64 once
#pragma unroll
    for (int i = 0; i < 4; ++i) {
        int ci = tid + i * 128;           // 0..511 chunks of 8 bf16
        int r = ci >> 3, cc = ci & 7;
        *(v8bf*)(&Qs[r * 72 + cc * 8]) = *(const v8bf*)(Qg + r * HDIM + cc * 8);
    }

    v8f oacc[4] = {};
    float mrow[8], lrow[8];
#pragma unroll
    for (int r = 0; r < 8; ++r) { mrow[r] = -1e30f; lrow[r] = 0.0f; }

    for (int kt = 0; kt < SEQ; kt += 64) {
        __syncthreads();   // Ks/Vt reads of previous tile done (also covers Qs 1st iter)
        // ---- stage K tile and transposed V tile -----------------------------
#pragma unroll
        for (int i = 0; i < 4; ++i) {
            int ci = tid + i * 128; int r = ci >> 3, cc = ci & 7;
            *(v8bf*)(&Ks[r * 72 + cc * 8]) = *(const v8bf*)(Kg + (size_t)(kt + r) * HDIM + cc * 8);
        }
#pragma unroll
        for (int i = 0; i < 4; ++i) {
            int ci = tid + i * 128; int r = ci >> 3, cc = ci & 7;   // key r, d-chunk cc
            v8bf v = *(const v8bf*)(Vg + (size_t)(kt + r) * HDIM + cc * 8);
#pragma unroll
            for (int j = 0; j < 8; ++j) Vt[(cc * 8 + j) * 72 + r] = v[j];
        }
        __syncthreads();

        // ---- S = Q K^T : 16 q rows x 64 keys (Q already carries 1/8 scale) --
        v8f sacc[4] = {};
#pragma unroll
        for (int ks = 0; ks < 64; ks += 32) {
            const int arow = wv_id * 16 + lh;
            v8bf alo = *(const v8bf*)(&Qs[arow * 72 + ks + hi16 * 8]);
            v8bf ahi = *(const v8bf*)(&Qs[arow * 72 + ks + hi16 * 8 + 16]);
            v16bf afrag = make_frag(alo, ahi);
#pragma unroll
            for (int nt = 0; nt < 4; ++nt) {
                const int krow = nt * 16 + lh;
                v8bf blo = *(const v8bf*)(&Ks[krow * 72 + ks + hi16 * 8]);
                v8bf bhi = *(const v8bf*)(&Ks[krow * 72 + ks + hi16 * 8 + 16]);
                v16bf bfrag = make_frag(blo, bhi);
                sacc[nt] = __builtin_amdgcn_wmma_f32_16x16x32_bf16(
                    false, afrag, false, bfrag, (short)0, sacc[nt], false, false);
            }
        }

        // ---- online softmax: rows live in 16-lane groups of the C layout ----
        float mnew[8], corr[8], rsum[8];
#pragma unroll
        for (int r = 0; r < 8; ++r) {
            float mx = fmaxf(fmaxf(sacc[0][r], sacc[1][r]), fmaxf(sacc[2][r], sacc[3][r]));
#pragma unroll
            for (int off = 1; off < 16; off <<= 1) mx = fmaxf(mx, __shfl_xor(mx, off, 32));
            mnew[r] = fmaxf(mrow[r], mx);
            corr[r] = __expf(mrow[r] - mnew[r]);
            mrow[r] = mnew[r];
            rsum[r] = 0.0f;
        }
        // probs -> bf16, stage through LDS to convert C-layout -> A-layout
#pragma unroll
        for (int nt = 0; nt < 4; ++nt) {
#pragma unroll
            for (int r = 0; r < 8; ++r) {
                float p = __expf(sacc[nt][r] - mnew[r]);
                rsum[r] += p;
                Ps[wv_id][(hi16 * 8 + r) * 72 + nt * 16 + lh] = (bf16)p;
            }
        }
#pragma unroll
        for (int r = 0; r < 8; ++r) {
            float s = rsum[r];
#pragma unroll
            for (int off = 1; off < 16; off <<= 1) s += __shfl_xor(s, off, 32);
            lrow[r] = lrow[r] * corr[r] + s;
#pragma unroll
            for (int nt = 0; nt < 4; ++nt) oacc[nt][r] *= corr[r];
        }

        // ---- O += P @ V  (DS ops are in-order per wave: no barrier needed) --
#pragma unroll
        for (int ks = 0; ks < 64; ks += 32) {
            v8bf alo = *(const v8bf*)(&Ps[wv_id][lh * 72 + ks + hi16 * 8]);
            v8bf ahi = *(const v8bf*)(&Ps[wv_id][lh * 72 + ks + hi16 * 8 + 16]);
            v16bf afrag = make_frag(alo, ahi);
#pragma unroll
            for (int nt = 0; nt < 4; ++nt) {
                const int drow = nt * 16 + lh;
                v8bf blo = *(const v8bf*)(&Vt[drow * 72 + ks + hi16 * 8]);
                v8bf bhi = *(const v8bf*)(&Vt[drow * 72 + ks + hi16 * 8 + 16]);
                v16bf bfrag = make_frag(blo, bhi);
                oacc[nt] = __builtin_amdgcn_wmma_f32_16x16x32_bf16(
                    false, afrag, false, bfrag, (short)0, oacc[nt], false, false);
            }
        }
    }

    // ---- epilogue: O /= l, store bf16 to [b, t, h*64+d] ---------------------
    const int b = bh >> 4, h = bh & 15;
#pragma unroll
    for (int nt = 0; nt < 4; ++nt) {
        const int d = nt * 16 + lh;
#pragma unroll
        for (int r = 0; r < 8; ++r) {
            const int t = q0 + wv_id * 16 + hi16 * 8 + r;
            const float val = oacc[nt][r] * __frcp_rn(lrow[r]);
            attn[((size_t)(b * SEQ + t) * D_MODEL) + h * HDIM + d] = (bf16)val;
        }
    }
}

// ---------------------------------------------------------------------------
// Kernel 3: output projection  out = attn(bf16) @ Wo + bo   (fp32 out)
// ---------------------------------------------------------------------------
__global__ __launch_bounds__(256)
void out_gemm_kernel(const bf16* __restrict__ A, const float* __restrict__ Wo,
                     const float* __restrict__ bo, float* __restrict__ out)
{
    __shared__ bf16 As[128 * 40];
    __shared__ bf16 Bt[64 * 40];

    const int tid    = threadIdx.x;
    const int lane   = tid & 31;
    const int wv_id  = tid >> 5;
    const int mBlock = blockIdx.y * 128;
    const int nBase  = blockIdx.x * 64;
    const int lh     = lane & 15;
    const int hi16   = lane >> 4;
    const int aK0    = hi16 * 8;
    const int bK0    = hi16 * 8;

    v8f acc[4] = {};

    for (int k0 = 0; k0 < D_MODEL; k0 += 32) {
#pragma unroll
        for (int i = 0; i < 2; ++i) {
            int idx = tid + i * 256;                 // 0..511 chunks of 8 bf16
            int r = idx >> 2, cc = idx & 3;
            *(v8bf*)(&As[r * 40 + cc * 8]) =
                *(const v8bf*)(A + (size_t)(mBlock + r) * D_MODEL + k0 + cc * 8);
        }
#pragma unroll
        for (int i = 0; i < 2; ++i) {
            int idx = tid + i * 256;
            int kr = idx >> 4, nc = idx & 15;
            v4f f = *(const v4f*)(Wo + (size_t)(k0 + kr) * D_MODEL + nBase + nc * 4);
#pragma unroll
            for (int j = 0; j < 4; ++j) Bt[(nc * 4 + j) * 40 + kr] = (bf16)f[j];
        }
        __syncthreads();

        const int arow = wv_id * 16 + lh;
        v8bf alo = *(const v8bf*)(&As[arow * 40 + aK0]);
        v8bf ahi = *(const v8bf*)(&As[arow * 40 + aK0 + 16]);
        v16bf afrag = make_frag(alo, ahi);
#pragma unroll
        for (int nt = 0; nt < 4; ++nt) {
            const int brow = nt * 16 + lh;
            v8bf blo = *(const v8bf*)(&Bt[brow * 40 + bK0]);
            v8bf bhi = *(const v8bf*)(&Bt[brow * 40 + bK0 + 16]);
            v16bf bfrag = make_frag(blo, bhi);
            acc[nt] = __builtin_amdgcn_wmma_f32_16x16x32_bf16(
                false, afrag, false, bfrag, (short)0, acc[nt], false, false);
        }
        __syncthreads();
    }

#pragma unroll
    for (int nt = 0; nt < 4; ++nt) {
        const int c = nBase + nt * 16 + lh;
        const float bval = bo[c];
#pragma unroll
        for (int r = 0; r < 8; ++r) {
            const int m = mBlock + wv_id * 16 + hi16 * 8 + r;
            out[(size_t)m * D_MODEL + c] = acc[nt][r] + bval;
        }
    }
}

// ---------------------------------------------------------------------------
extern "C" void kernel_launch(void* const* d_in, const int* in_sizes, int n_in,
                              void* d_out, int out_size, void* d_ws, size_t ws_size,
                              hipStream_t stream)
{
    const float* x  = (const float*)d_in[0];
    const float* Wq = (const float*)d_in[1];
    const float* bq = (const float*)d_in[2];
    const float* Wk = (const float*)d_in[3];
    const float* bk = (const float*)d_in[4];
    const float* Wv = (const float*)d_in[5];
    const float* bv = (const float*)d_in[6];
    const float* Wo = (const float*)d_in[7];
    const float* bo = (const float*)d_in[8];
    float* out = (float*)d_out;

    // workspace: Q | K | V | attn, each 4096*1024 bf16 (8 MB) -> 32 MB total
    bf16* qkv  = (bf16*)d_ws;
    bf16* attn = qkv + (size_t)3 * ROWS * D_MODEL;

    dim3 g1(D_MODEL / 64, ROWS / 128, 3);
    qkv_gemm_kernel<<<g1, 256, 0, stream>>>(x, Wq, bq, Wk, bk, Wv, bv, qkv);

    dim3 g2(SEQ / 64, BATCH * NHEAD);
    flash_attn_kernel<<<g2, 128, 0, stream>>>(
        qkv,
        qkv + (size_t)ROWS * D_MODEL,
        qkv + (size_t)2 * ROWS * D_MODEL,
        attn);

    dim3 g3(D_MODEL / 64, ROWS / 128);
    out_gemm_kernel<<<g3, 256, 0, stream>>>(attn, Wo, bo, out);
}
